// Decoder_23819888623930
// MI455X (gfx1250) — compile-verified, matrix operand order checked by reading, not run on hardware
//
#include <hip/hip_runtime.h>
#include <hip/hip_bf16.h>

// Sizes from the reference
#define V_  32000
#define E_  512
#define H_  1024
#define L_  2
#define B_  32
#define T_  32
#define S_  128

typedef __attribute__((ext_vector_type(16))) __bf16 v16bf;
typedef __attribute__((ext_vector_type(8)))  float  v8f;
typedef unsigned short u16_t;

// CDNA5 async LDS path (guarded: falls back to load+ds_store if absent)
#if defined(__has_builtin)
# if __has_builtin(__builtin_amdgcn_global_load_async_to_lds_b128)
#  define USE_ASYNC_LDS 1
# endif
#endif
#ifndef USE_ASYNC_LDS
# define USE_ASYNC_LDS 0
#endif

// Exact pointer types the async-LDS builtin expects (from hipcc diagnostic):
// param0: vector_size(16) int, address_space(1); param1: same pointee, AS3.
typedef int b128v __attribute__((vector_size(16)));
typedef __attribute__((address_space(1))) b128v* gaddr128_t;
typedef __attribute__((address_space(3))) b128v* laddr128_t;

union FragBF { uint4 q[2]; v16bf v; };

__device__ __forceinline__ u16_t f2bf(float x) {
    unsigned u = __float_as_uint(x);
    u = u + 0x7FFFu + ((u >> 16) & 1u);   // round-to-nearest-even
    return (u16_t)(u >> 16);
}
__device__ __forceinline__ float sigf(float x) { return 1.0f / (1.0f + __expf(-x)); }

// Load one 16x32 bf16 WMMA fragment row: row_kk points at X[row*K + kk]
// lane layout (ISA 16-bit A 16x32): elems 0..7 -> k = kbase..kbase+7,
// elems 8..15 -> k = 16+kbase..16+kbase+7, kbase = (lane>>4)*8.
__device__ __forceinline__ v16bf load_frag(const u16_t* row_kk, int kbase) {
    FragBF f;
    f.q[0] = *reinterpret_cast<const uint4*>(row_kk + kbase);
    f.q[1] = *reinterpret_cast<const uint4*>(row_kk + kbase + 16);
    return f.v;
}

// ---------------- one-time prep kernels ----------------

__global__ void prep_keys_kernel(const float* __restrict__ enc, float* __restrict__ keys) {
    int i = blockIdx.x * 256 + threadIdx.x;              // over B*S*H
    if (i >= B_ * S_ * H_) return;
    int h = i & (H_ - 1);
    int bs = i >> 10;
    keys[i] = enc[(size_t)bs * (2 * H_) + h] + enc[(size_t)bs * (2 * H_) + H_ + h];
}

__global__ void cvt_bf16_kernel(const float* __restrict__ src, u16_t* __restrict__ dst, long n) {
    long i = (long)blockIdx.x * 256 + threadIdx.x;
    if (i < n) dst[i] = f2bf(src[i]);
}

__global__ void build_wcat_kernel(const float* __restrict__ Wih, int Kih,
                                  const float* __restrict__ Whh,
                                  u16_t* __restrict__ out, int K) {
    long i = (long)blockIdx.x * 256 + threadIdx.x;       // over 4096*K
    long total = (long)(4 * H_) * K;
    if (i >= total) return;
    int k = (int)(i % K);
    int r = (int)(i / K);
    float v = (k < Kih) ? Wih[(long)r * Kih + k] : Whh[(long)r * H_ + (k - Kih)];
    out[i] = f2bf(v);
}

__global__ void combine_bias_kernel(const float* __restrict__ a, const float* __restrict__ b,
                                    float* __restrict__ o) {
    int i = blockIdx.x * 256 + threadIdx.x;
    if (i < 4 * H_) o[i] = a[i] + b[i];
}

__global__ void gather_emb_kernel(const int* __restrict__ tgt, const float* __restrict__ emb,
                                  u16_t* __restrict__ out) {
    int i = blockIdx.x * 256 + threadIdx.x;              // over B*T*E
    if (i >= B_ * T_ * E_) return;
    int e = i & (E_ - 1);
    int bt = i >> 9;
    int tok = tgt[bt];
    float v = (tok == 0) ? 0.0f : emb[(size_t)tok * E_ + e];   // padding_idx == 0
    out[i] = f2bf(v);
}

// ---------------- per-step kernels ----------------

// One block per batch row. Computes masked softmax attention over keys,
// then packs all bf16 concat inputs for this step's GEMMs.
__global__ void __launch_bounds__(128)
attn_kernel(const float* __restrict__ keys,      // [B,S,H]
            const int*   __restrict__ src_mask,  // [B,S]
            const float* __restrict__ h_state,   // [2,B,H]
            const u16_t* __restrict__ emb_bf,    // [B,T,E]
            u16_t* __restrict__ z0,              // [B,2560] = [emb | ctx | h0prev]
            u16_t* __restrict__ z1,              // [B,2048] = [h0new | h1prev]
            u16_t* __restrict__ zfc,             // [B,2048] = [h1new | ctx]
            int t) {
    __shared__ float q_sh[H_];
    __shared__ float a_sh[S_];
    __shared__ float red[S_];
    const int b   = blockIdx.x;
    const int tid = threadIdx.x;
    const float* q   = h_state + ((size_t)(1 * B_ + b)) * H_;   // top layer hidden
    const float* h0p = h_state + ((size_t)(0 * B_ + b)) * H_;

    for (int h = tid; h < H_; h += 128) q_sh[h] = q[h];
    __syncthreads();

    // scores (one s per thread)
    const float* krow = keys + ((size_t)b * S_ + tid) * H_;
    float s = 0.0f;
    for (int h = 0; h < H_; ++h) s += krow[h] * q_sh[h];
    if (src_mask[b * S_ + tid] == 0) s = -1e9f;

    red[tid] = s;
    __syncthreads();
    for (int off = 64; off > 0; off >>= 1) {
        if (tid < off) red[tid] = fmaxf(red[tid], red[tid + off]);
        __syncthreads();
    }
    float mx = red[0];
    __syncthreads();
    float e = __expf(s - mx);
    red[tid] = e;
    __syncthreads();
    for (int off = 64; off > 0; off >>= 1) {
        if (tid < off) red[tid] += red[tid + off];
        __syncthreads();
    }
    float inv = 1.0f / red[0];
    __syncthreads();
    a_sh[tid] = e * inv;
    __syncthreads();

    // ctx + pack bf16 outputs
    for (int h = tid; h < H_; h += 128) {
        const float* kcol = keys + (size_t)b * S_ * H_ + h;
        float c = 0.0f;
        for (int si = 0; si < S_; ++si) c += a_sh[si] * kcol[(size_t)si * H_];
        u16_t cb = f2bf(c);
        z0[b * 2560 + E_ + h]   = cb;                 // ctx slice of z0
        zfc[b * 2048 + H_ + h]  = cb;                 // ctx slice of zfc
        z0[b * 2560 + 1536 + h] = f2bf(h0p[h]);       // h0prev slice of z0
        z1[b * 2048 + H_ + h]   = f2bf(q_sh[h]);      // h1prev slice of z1
    }
    for (int e2 = tid; e2 < E_; e2 += 128)
        z0[b * 2560 + e2] = emb_bf[((size_t)b * T_ + t) * E_ + e2];
}

// One wave per block. Block hb covers 16 h-columns; the 4 N-subtiles are the
// SAME h across the i/f/g/o gate blocks, so the LSTM cell update fuses in-register.
__global__ void __launch_bounds__(32)
lstm_step_kernel(const u16_t* __restrict__ Z, int K,       // [32, K] bf16
                 const u16_t* __restrict__ W,              // [4096, K] bf16
                 const float* __restrict__ bias,           // [4096]
                 float* __restrict__ c_io,                 // [32, 1024]
                 float* __restrict__ h_out,                // [32, 1024]
                 u16_t* __restrict__ zout, int zpitch) {   // bf16 h_new slice
    const int lane  = threadIdx.x & 31;
    const int hb    = blockIdx.x * 16;
    const int kbase = (lane >> 4) * 8;
    const int nl    = lane & 15;

    v8f acc[2][4] = {};
    for (int kk = 0; kk < K; kk += 32) {
        v16bf a0 = load_frag(Z + (size_t)nl * K + kk, kbase);
        v16bf a1 = load_frag(Z + (size_t)(16 + nl) * K + kk, kbase);
#pragma unroll
        for (int j = 0; j < 4; ++j) {
            v16bf bf = load_frag(W + (size_t)(j * H_ + hb + nl) * K + kk, kbase);
            acc[0][j] = __builtin_amdgcn_wmma_f32_16x16x32_bf16(
                false, a0, false, bf, (short)0, acc[0][j], false, false);
            acc[1][j] = __builtin_amdgcn_wmma_f32_16x16x32_bf16(
                false, a1, false, bf, (short)0, acc[1][j], false, false);
        }
    }

    const int h = hb + nl;
    const float bi  = bias[0 * H_ + h];
    const float bf_ = bias[1 * H_ + h];
    const float bg  = bias[2 * H_ + h];
    const float bo  = bias[3 * H_ + h];
#pragma unroll
    for (int mt = 0; mt < 2; ++mt) {
#pragma unroll
        for (int v = 0; v < 8; ++v) {
            int m = mt * 16 + ((lane >> 4) << 3) + v;     // C/D layout: M=v(+8), N=lane&15
            float ig = sigf(acc[mt][0][v] + bi);
            float fg = sigf(acc[mt][1][v] + bf_);
            float gg = tanhf(acc[mt][2][v] + bg);
            float og = sigf(acc[mt][3][v] + bo);
            float cp = c_io[m * H_ + h];
            float cn = fg * cp + ig * gg;
            float hn = og * tanhf(cn);
            c_io[m * H_ + h]  = cn;
            h_out[m * H_ + h] = hn;
            zout[m * zpitch + h] = f2bf(hn);
        }
    }
}

// FC: 4 waves/block, each wave 2 M-tiles x 4 N-tiles (64 cols); 125 blocks x 256
// cols = exactly 32000. The activation matrix zfc (32x2048 bf16) is staged once
// per block into LDS (row pitch 2056 -> +16B padding, conflict-free b128 reads),
// cutting redundant L2 A-traffic from ~64 MB/step to 16 MB/step; fc_w stays the
// single 131 MB bf16 L2-resident stream.
#define FC_LDS_PITCH 2056
__global__ void __launch_bounds__(128)
fc_kernel(const u16_t* __restrict__ Zfc,     // [32, 2048] bf16
          const u16_t* __restrict__ Wfc,     // [32000, 2048] bf16
          const float* __restrict__ fcb,     // [32000]
          float* __restrict__ out, int t) {
    __shared__ u16_t sh[32 * FC_LDS_PITCH];  // 131,584 B of the 320 KB WGP LDS
    const int tid = threadIdx.x;

    // Stage A into LDS: 32 rows x 256 uint4 chunks
#if USE_ASYNC_LDS
    for (int idx = tid; idx < 32 * 256; idx += 128) {
        int m = idx >> 8, c = idx & 255;
        __builtin_amdgcn_global_load_async_to_lds_b128(
            (gaddr128_t)(Zfc + m * 2048 + c * 8),
            (laddr128_t)(sh + m * FC_LDS_PITCH + c * 8),
            0, 0);
    }
# if defined(__has_builtin) && __has_builtin(__builtin_amdgcn_s_wait_asynccnt)
    __builtin_amdgcn_s_wait_asynccnt(0);
# else
    asm volatile("s_wait_asynccnt 0x0" ::: "memory");
# endif
    __syncthreads();
#else
    for (int idx = tid; idx < 32 * 256; idx += 128) {
        int m = idx >> 8, c = idx & 255;
        *reinterpret_cast<uint4*>(sh + m * FC_LDS_PITCH + c * 8) =
            *reinterpret_cast<const uint4*>(Zfc + m * 2048 + c * 8);
    }
    __syncthreads();
#endif

    const int lane  = tid & 31;
    const int wave  = tid >> 5;
    const int nb    = blockIdx.x * 256 + wave * 64;
    const int kbase = (lane >> 4) * 8;
    const int nl    = lane & 15;
    const int K     = 2 * H_;

    v8f acc[2][4] = {};
    for (int kk = 0; kk < K; kk += 32) {
        v16bf a0 = load_frag(sh + nl * FC_LDS_PITCH + kk, kbase);          // ds_load_b128
        v16bf a1 = load_frag(sh + (16 + nl) * FC_LDS_PITCH + kk, kbase);
#pragma unroll
        for (int nt = 0; nt < 4; ++nt) {
            v16bf b = load_frag(Wfc + (size_t)(nb + nt * 16 + nl) * K + kk, kbase);
            acc[0][nt] = __builtin_amdgcn_wmma_f32_16x16x32_bf16(
                false, a0, false, b, (short)0, acc[0][nt], false, false);
            acc[1][nt] = __builtin_amdgcn_wmma_f32_16x16x32_bf16(
                false, a1, false, b, (short)0, acc[1][nt], false, false);
        }
    }

#pragma unroll
    for (int nt = 0; nt < 4; ++nt) {
        int n = nb + nt * 16 + nl;
        float bv = fcb[n];
#pragma unroll
        for (int mt = 0; mt < 2; ++mt) {
#pragma unroll
            for (int v = 0; v < 8; ++v) {
                int m = mt * 16 + ((lane >> 4) << 3) + v;
                out[(size_t)m * T_ * V_ + (size_t)t * V_ + n] = acc[mt][nt][v] + bv;
            }
        }
    }
}

// ---------------- host ----------------

extern "C" void kernel_launch(void* const* d_in, const int* in_sizes, int n_in,
                              void* d_out, int out_size, void* d_ws, size_t ws_size,
                              hipStream_t stream) {
    const int*   tgt_in   = (const int*)d_in[0];
    const int*   src_mask = (const int*)d_in[1];
    const float* h0       = (const float*)d_in[2];
    const float* c0       = (const float*)d_in[3];
    const float* enc      = (const float*)d_in[4];
    const float* emb      = (const float*)d_in[5];
    const float* W_ih0    = (const float*)d_in[6];
    const float* W_hh0    = (const float*)d_in[7];
    const float* b_ih0    = (const float*)d_in[8];
    const float* b_hh0    = (const float*)d_in[9];
    const float* W_ih1    = (const float*)d_in[10];
    const float* W_hh1    = (const float*)d_in[11];
    const float* b_ih1    = (const float*)d_in[12];
    const float* b_hh1    = (const float*)d_in[13];
    const float* fc_w     = (const float*)d_in[14];
    const float* fc_b     = (const float*)d_in[15];
    float* out = (float*)d_out;

    char* ws = (char*)d_ws;
    size_t off = 0;
    auto carve = [&](size_t bytes) -> void* {
        void* p = ws + off;
        off = (off + bytes + 255) & ~(size_t)255;
        return p;
    };
    float* keys   = (float*)carve((size_t)B_ * S_ * H_ * 4);          // 16 MB
    u16_t* fcw    = (u16_t*)carve((size_t)V_ * 2 * H_ * 2);           // 131 MB (L2-resident)
    u16_t* wc0    = (u16_t*)carve((size_t)4 * H_ * 2560 * 2);         // 21 MB
    u16_t* wc1    = (u16_t*)carve((size_t)4 * H_ * 2048 * 2);         // 16.8 MB
    u16_t* embb   = (u16_t*)carve((size_t)B_ * T_ * E_ * 2);          // 1 MB
    float* hstate = (float*)carve((size_t)L_ * B_ * H_ * 4);
    float* cstate = (float*)carve((size_t)L_ * B_ * H_ * 4);
    u16_t* z0     = (u16_t*)carve((size_t)B_ * 2560 * 2);
    u16_t* z1     = (u16_t*)carve((size_t)B_ * 2048 * 2);
    u16_t* zfc    = (u16_t*)carve((size_t)B_ * 2048 * 2);
    float* bc0    = (float*)carve((size_t)4 * H_ * 4);
    float* bc1    = (float*)carve((size_t)4 * H_ * 4);

    // One-time prep
    prep_keys_kernel<<<(B_ * S_ * H_) / 256, 256, 0, stream>>>(enc, keys);
    {
        long n = (long)V_ * 2 * H_;
        cvt_bf16_kernel<<<(int)((n + 255) / 256), 256, 0, stream>>>(fc_w, fcw, n);
    }
    {
        long n = (long)4 * H_ * 2560;
        build_wcat_kernel<<<(int)((n + 255) / 256), 256, 0, stream>>>(W_ih0, E_ + H_, W_hh0, wc0, 2560);
    }
    {
        long n = (long)4 * H_ * 2048;
        build_wcat_kernel<<<(int)((n + 255) / 256), 256, 0, stream>>>(W_ih1, H_, W_hh1, wc1, 2048);
    }
    combine_bias_kernel<<<16, 256, 0, stream>>>(b_ih0, b_hh0, bc0);
    combine_bias_kernel<<<16, 256, 0, stream>>>(b_ih1, b_hh1, bc1);
    gather_emb_kernel<<<(B_ * T_ * E_) / 256, 256, 0, stream>>>(tgt_in, emb, embb);
    (void)hipMemcpyAsync(hstate, h0, (size_t)L_ * B_ * H_ * 4, hipMemcpyDeviceToDevice, stream);
    (void)hipMemcpyAsync(cstate, c0, (size_t)L_ * B_ * H_ * 4, hipMemcpyDeviceToDevice, stream);

    // Sequential timestep recurrence: stream ordering gives device-wide sync
    for (int t = 0; t < T_; ++t) {
        attn_kernel<<<B_, 128, 0, stream>>>(keys, src_mask, hstate, embb, z0, z1, zfc, t);
        lstm_step_kernel<<<H_ / 16, 32, 0, stream>>>(z0, 2560, wc0, bc0,
                                                     cstate, hstate, z1, 2048);
        lstm_step_kernel<<<H_ / 16, 32, 0, stream>>>(z1, 2048, wc1, bc1,
                                                     cstate + B_ * H_, hstate + B_ * H_, zfc, 2048);
        fc_kernel<<<V_ / 256, 128, 0, stream>>>(zfc, fcw, fc_b, out, t);
    }
}
